// AttnFuser_6837587935839
// MI455X (gfx1250) — matrix operand order repaired and checked by reading, not compiled, not run on hardware
//
#include <hip/hip_runtime.h>
#include <hip/hip_bf16.h>

// ---------------------------------------------------------------------------
// Types for CDNA5 WMMA (wave32): A/B = 16 bf16 per lane, C/D = 8 f32 per lane.
// ---------------------------------------------------------------------------
typedef __attribute__((ext_vector_type(16))) __bf16 v16bf;
typedef __attribute__((ext_vector_type(8)))  __bf16 v8bf;
typedef __attribute__((ext_vector_type(8)))  float  v8f;
typedef __attribute__((ext_vector_type(4)))  unsigned int v4u;
typedef __attribute__((ext_vector_type(8)))  int    v8i;
typedef __attribute__((ext_vector_type(4)))  int    v4i;

__device__ __forceinline__ v16bf pack16(v8bf lo, v8bf hi) {
  v16bf r;
#pragma unroll
  for (int i = 0; i < 8; ++i) { r[i] = lo[i]; r[8 + i] = hi[i]; }
  return r;
}

__device__ __forceinline__ v8f wmma_bf16(v16bf a, v16bf b, v8f c) {
  // D = A(16x32 bf16) * B(32x16 bf16) + C(16x16 f32)
  return __builtin_amdgcn_wmma_f32_16x16x32_bf16(false, a, false, b, (short)0, c,
                                                 false, false);
}

// ---------------------------------------------------------------------------
// Elementwise f32 -> bf16 conversion (weights + activations).
// ---------------------------------------------------------------------------
__global__ __launch_bounds__(256) void cvt_f32_bf16_kernel(
    const float* __restrict__ in, __bf16* __restrict__ out, int n) {
  int i = (blockIdx.x * 256 + threadIdx.x) * 4;
  if (i + 3 < n) {
    float4 v = *(const float4*)(in + i);
    out[i + 0] = (__bf16)v.x;
    out[i + 1] = (__bf16)v.y;
    out[i + 2] = (__bf16)v.z;
    out[i + 3] = (__bf16)v.w;
  }
}

// ---------------------------------------------------------------------------
// Tiled WMMA GEMM: out[Mr,Nc] = act( A[Mr,Kd]*Bw[Kd,Nc] + bias ) (+ resid)
// Block = 256 threads (8 waves). Block tile 64x64. Wave (wr 0..3, wc 0..1)
// computes a 16x32 strip: one A fragment reused across two B fragments.
// A tile (64x32, row-major) is DMA'd into LDS by the Tensor Data Mover
// (one tensor_load_to_lds per K-step, TENSORcnt-tracked); B is staged
// K-major by the threads so each lane's B fragment is two ds_load_b128.
// ACT: 0 = none, 1 = exact GELU.  OBF: 0 = f32 out, 1 = bf16 out.
// ---------------------------------------------------------------------------
template <int ACT, int OBF>
__global__ __launch_bounds__(256) void gemm_bf16_kernel(
    const __bf16* __restrict__ A, const __bf16* __restrict__ Bw,
    const float* __restrict__ bias, const float* __restrict__ resid,
    void* __restrict__ outp, int Mr, int Kd, int Nc) {
  __shared__ __bf16 Asm[64][32];   // [m][k]   (TDM destination, row-major)
  __shared__ __bf16 Bt[64][32];    // [n][k]   (transposed)

  const int tid  = threadIdx.x;
  const int lane = tid & 31;
  const int wave = tid >> 5;        // 0..7
  const int wr   = wave & 3;        // row tile
  const int wc   = wave >> 2;       // col pair
  const int half = lane >> 4;       // 0/1
  const int l16  = lane & 15;
  const int m0   = blockIdx.x * 64;
  const int n0   = blockIdx.y * 64;

  // LDS byte offset of Asm (flat LDS address low 32 bits == LDS offset).
  const unsigned lds_a = (unsigned)(uintptr_t)(&Asm[0][0]);

  v8f acc0 = {}; v8f acc1 = {};

  for (int k0 = 0; k0 < Kd; k0 += 32) {
    // --- A tile via Tensor Data Mover: 64 rows x 32 bf16, row-major ---
    if (tid == 0) {
      unsigned long long ga =
          (unsigned long long)(uintptr_t)(A + (size_t)m0 * Kd + k0);
      v4u g0;
      g0[0] = 1u;                                   // count=1 (valid user D#)
      g0[1] = lds_a;                                // lds_addr
      g0[2] = (unsigned)(ga & 0xffffffffu);         // global_addr[31:0]
      g0[3] = (unsigned)((ga >> 32) & 0x01ffffffu)  // global_addr[56:32]
              | (2u << 30);                         // type = 2 ("image")
      v8i g1;
      g1[0] = (1 << 16);                            // data_size=1 -> 2 bytes
      g1[1] = (int)(((unsigned)Kd & 0xffffu) << 16);        // tensor_dim0 lo16
      g1[2] = (int)((((unsigned)Kd >> 16) & 0xffffu)        // tensor_dim0 hi16
              | (((unsigned)Mr & 0xffffu) << 16));          // tensor_dim1 lo16
      g1[3] = (int)((((unsigned)Mr >> 16) & 0xffffu)        // tensor_dim1 hi16
              | (32u << 16));                               // tile_dim0 = 32
      g1[4] = 64;                                   // tile_dim1=64, tile_dim2=0
      g1[5] = Kd;                                   // tensor_dim0_stride[31:0]
      g1[6] = 0;                                    // stride hi / dim1_stride lo
      g1[7] = 0;
      v4i gz4 = {0, 0, 0, 0};
      v8i gz8 = {0, 0, 0, 0, 0, 0, 0, 0};
      __builtin_amdgcn_tensor_load_to_lds(g0, g1, gz4, gz4, gz8, 0);
    }
    // --- stage B tile 32x64, transposed into Bt[n][k] ---
    {
      int k  = tid >> 3;                  // 0..31
      int n8 = (tid & 7) * 8;
      v8bf bv = *(const v8bf*)(Bw + (size_t)(k0 + k) * Nc + n0 + n8);
#pragma unroll
      for (int i = 0; i < 8; ++i) Bt[n8 + i][k] = bv[i];
      if (k0 + 32 < Kd)
        __builtin_prefetch(Bw + (size_t)(k0 + 32 + k) * Nc + n0 + n8, 0, 1);
    }
    if (tid < 32) __builtin_amdgcn_s_wait_tensorcnt(0);  // wave0 issued the TDM
    __syncthreads();

    // A fragment: lane holds row (wr*16+l16), K block half*16..half*16+15
    v16bf af;
    {
      int row = wr * 16 + l16;
      int kb  = half * 16;
      af = pack16(*(const v8bf*)(&Asm[row][kb]),
                  *(const v8bf*)(&Asm[row][kb + 8]));
    }
    // B fragments: lane holds column, K block half*16..half*16+15
#pragma unroll
    for (int t = 0; t < 2; ++t) {
      int ncol = wc * 32 + t * 16 + l16;
      int kb   = half * 16;
      v16bf bf_ = pack16(*(const v8bf*)(&Bt[ncol][kb]),
                         *(const v8bf*)(&Bt[ncol][kb + 8]));
      if (t == 0) acc0 = wmma_bf16(af, bf_, acc0);
      else        acc1 = wmma_bf16(af, bf_, acc1);
    }
    __syncthreads();
  }

  // --- epilogue: bias (+GELU) (+residual), store f32 or bf16 ---
#pragma unroll
  for (int t = 0; t < 2; ++t) {
    v8f acc = t ? acc1 : acc0;
    int col = n0 + wc * 32 + t * 16 + l16;
    float bcol = bias[col];
#pragma unroll
    for (int i = 0; i < 8; ++i) {
      int row = m0 + wr * 16 + half * 8 + i;
      float v = acc[i] + bcol;
      if (ACT == 1) v = 0.5f * v * (1.0f + erff(v * 0.70710678118654752f));
      if (resid) v += resid[(size_t)row * Nc + col];
      if (OBF) ((__bf16*)outp)[(size_t)row * Nc + col] = (__bf16)v;
      else     ((float*)outp)[(size_t)row * Nc + col]  = v;
    }
  }
}

// ---------------------------------------------------------------------------
// Per-head LayerNorm + RoPE. One wave per (b,s,h) row of d=64 (2 vals/lane).
// Reads f32 [B,S,H*64], writes bf16 [B,H,S,64]. `scale` folds 1/sqrt(d) for Q.
// ---------------------------------------------------------------------------
__global__ __launch_bounds__(256) void lnrope_kernel(
    const float* __restrict__ X, const int* __restrict__ pos,
    const float* __restrict__ g, const float* __restrict__ bb,
    __bf16* __restrict__ out, int S, float scale) {
  int wv   = (blockIdx.x * 256 + threadIdx.x) >> 5;
  int lane = threadIdx.x & 31;
  int h = wv % 16;
  int s = (wv / 16) % S;
  int b = wv / (16 * S);

  const float* x = X + (((size_t)b * S + s) * 1024) + h * 64;
  float e0 = x[lane], e1 = x[lane + 32];
  float sum = e0 + e1, sq = e0 * e0 + e1 * e1;
#pragma unroll
  for (int m = 16; m; m >>= 1) {
    sum += __shfl_xor(sum, m, 32);
    sq  += __shfl_xor(sq,  m, 32);
  }
  float mu   = sum * (1.0f / 64.0f);
  float var  = sq * (1.0f / 64.0f) - mu * mu;
  float rstd = rsqrtf(var + 1e-5f);
  e0 = (e0 - mu) * rstd * g[lane]      + bb[lane];
  e1 = (e1 - mu) * rstd * g[lane + 32] + bb[lane + 32];

  float p    = (float)pos[(size_t)b * S + s];
  // inv_freq = 10000^(-lane/32) = exp2(-lane * log2(10000)/32)
  float invf = exp2f(-0.41524101186092029f * (float)lane);
  float ang  = p * invf;
  float sn, cs;
  __sincosf(ang, &sn, &cs);
  float o0 = (e0 * cs - e1 * sn) * scale;
  float o1 = (e0 * sn + e1 * cs) * scale;

  __bf16* op = out + ((((size_t)b * 16 + h) * S + s) * 64);
  op[lane]      = (__bf16)o0;
  op[lane + 32] = (__bf16)o1;
}

// ---------------------------------------------------------------------------
// Flash attention. One wave per 16 query rows of one (b,h). Streams K/V in
// 32-key chunks. S = Q*K^T (4 wmma), online softmax (cross-lane row max/sum),
// P restaged via LDS into A layout, V staged d-major in LDS, O += P*V (4 wmma).
// Epilogue: O/l + residual(query) -> o f32 [B,N,1024].
// ---------------------------------------------------------------------------
__global__ __launch_bounds__(32) void flash_attn_kernel(
    const __bf16* __restrict__ Qb,   // [B,H,N,64] (pre-scaled by 1/8)
    const __bf16* __restrict__ Kb,   // [B,H,M,64]
    const __bf16* __restrict__ Vb,   // [B,M,H*64]
    const float* __restrict__ resid, // query [B,N,1024]
    float* __restrict__ O,           // [B,N,1024]
    int Nq, int Mk) {
  __shared__ __bf16 Pls[16][32];     // P chunk, row-major
  __shared__ __bf16 Vt[64][32];      // V chunk, [d][m]

  const int lane = threadIdx.x;
  const int half = lane >> 4, l16 = lane & 15;
  const int n0 = blockIdx.x * 16;
  const int h  = blockIdx.y;
  const int b  = blockIdx.z;

  const __bf16* Qp = Qb + (((size_t)b * 16 + h) * Nq + n0) * 64;
  const __bf16* Kp = Kb + (((size_t)b * 16 + h) * Mk) * 64;
  const __bf16* Vp = Vb + ((size_t)b * Mk) * 1024 + h * 64;

  // Q fragments (contraction d=64 -> two 16x32 A fragments)
  int kb8 = half * 16;
  const __bf16* qr = Qp + l16 * 64;
  v16bf aq0 = pack16(*(const v8bf*)(qr + kb8),      *(const v8bf*)(qr + kb8 + 8));
  v16bf aq1 = pack16(*(const v8bf*)(qr + 32 + kb8), *(const v8bf*)(qr + 32 + kb8 + 8));

  float mrow[8], lrow[8];
  v8f o0 = {}, o1 = {}, o2 = {}, o3 = {};
#pragma unroll
  for (int i = 0; i < 8; ++i) { mrow[i] = -1e30f; lrow[i] = 0.0f; }

  for (int j0 = 0; j0 < Mk; j0 += 32) {
    // --- stage V chunk (32 x 64) d-major: Vt[d][m] ---
    {
      const __bf16* vr = Vp + (size_t)(j0 + lane) * 1024;
#pragma unroll
      for (int q = 0; q < 8; ++q) {
        v8bf vv = *(const v8bf*)(vr + q * 8);
#pragma unroll
        for (int i = 0; i < 8; ++i) Vt[q * 8 + i][lane] = vv[i];
      }
    }
    // --- scores: two 16x16 tiles, each 2 wmma over d ---
    v8f s0 = {}, s1 = {};
#pragma unroll
    for (int t = 0; t < 2; ++t) {
      const __bf16* kr = Kp + (size_t)(j0 + t * 16 + l16) * 64;
      v16bf bk0 = pack16(*(const v8bf*)(kr + half * 16),
                         *(const v8bf*)(kr + half * 16 + 8));
      v16bf bk1 = pack16(*(const v8bf*)(kr + 32 + half * 16),
                         *(const v8bf*)(kr + 32 + half * 16 + 8));
      v8f st = {};
      st = wmma_bf16(aq0, bk0, st);
      st = wmma_bf16(aq1, bk1, st);
      if (t == 0) s0 = st; else s1 = st;
    }
    // --- online softmax (row r = half*8+i lives across 16 lanes) ---
#pragma unroll
    for (int i = 0; i < 8; ++i) {
      float mx = fmaxf(s0[i], s1[i]);
#pragma unroll
      for (int msk = 1; msk < 16; msk <<= 1) mx = fmaxf(mx, __shfl_xor(mx, msk, 32));
      float nm = fmaxf(mrow[i], mx);
      float alpha = __expf(mrow[i] - nm);
      mrow[i] = nm;
      float p0 = __expf(s0[i] - nm);
      float p1 = __expf(s1[i] - nm);
      float ps = p0 + p1;
#pragma unroll
      for (int msk = 1; msk < 16; msk <<= 1) ps += __shfl_xor(ps, msk, 32);
      lrow[i] = lrow[i] * alpha + ps;
      o0[i] *= alpha; o1[i] *= alpha; o2[i] *= alpha; o3[i] *= alpha;
      int row = half * 8 + i;
      Pls[row][l16]      = (__bf16)p0;
      Pls[row][16 + l16] = (__bf16)p1;
    }
    asm volatile("s_wait_dscnt 0" ::: "memory");  // P stores visible before reload

    // --- P as A fragment (16x32), V tiles as B fragments, O += P*V ---
    v16bf ap = pack16(*(const v8bf*)(&Pls[l16][kb8]),
                      *(const v8bf*)(&Pls[l16][kb8 + 8]));
#pragma unroll
    for (int t = 0; t < 4; ++t) {
      int dd = t * 16 + l16;
      int kb = half * 16;
      v16bf bv = pack16(*(const v8bf*)(&Vt[dd][kb]),
                        *(const v8bf*)(&Vt[dd][kb + 8]));
      if      (t == 0) o0 = wmma_bf16(ap, bv, o0);
      else if (t == 1) o1 = wmma_bf16(ap, bv, o1);
      else if (t == 2) o2 = wmma_bf16(ap, bv, o2);
      else             o3 = wmma_bf16(ap, bv, o3);
    }
  }

  // --- epilogue: normalize + residual ---
  float* Op       = O     + ((size_t)b * Nq + n0) * 1024 + h * 64;
  const float* Rp = resid + ((size_t)b * Nq + n0) * 1024 + h * 64;
#pragma unroll
  for (int i = 0; i < 8; ++i) {
    int row = half * 8 + i;
    float inv = 1.0f / lrow[i];
    Op[(size_t)row * 1024 + l16]      = o0[i] * inv + Rp[(size_t)row * 1024 + l16];
    Op[(size_t)row * 1024 + 16 + l16] = o1[i] * inv + Rp[(size_t)row * 1024 + 16 + l16];
    Op[(size_t)row * 1024 + 32 + l16] = o2[i] * inv + Rp[(size_t)row * 1024 + 32 + l16];
    Op[(size_t)row * 1024 + 48 + l16] = o3[i] * inv + Rp[(size_t)row * 1024 + 48 + l16];
  }
}

// ---------------------------------------------------------------------------
// LayerNorm over D=1024 (FFN pre-norm), one block (256 thr) per row, bf16 out.
// ---------------------------------------------------------------------------
__global__ __launch_bounds__(256) void ln_d_kernel(
    const float* __restrict__ X, const float* __restrict__ g,
    const float* __restrict__ bb, __bf16* __restrict__ out, int D) {
  __shared__ float red[18];
  size_t row = blockIdx.x;
  const float* x = X + row * D;
  float sum = 0.0f, sq = 0.0f;
  for (int i = threadIdx.x; i < D; i += 256) {
    float v = x[i];
    sum += v; sq += v * v;
  }
#pragma unroll
  for (int m = 16; m; m >>= 1) {
    sum += __shfl_xor(sum, m, 32);
    sq  += __shfl_xor(sq,  m, 32);
  }
  int wave = threadIdx.x >> 5, lane = threadIdx.x & 31;
  if (lane == 0) { red[wave] = sum; red[wave + 8] = sq; }
  __syncthreads();
  if (threadIdx.x == 0) {
    float S = 0.0f, Q = 0.0f;
    for (int i = 0; i < 8; ++i) { S += red[i]; Q += red[i + 8]; }
    float mu  = S / (float)D;
    float var = Q / (float)D - mu * mu;
    red[16] = mu;
    red[17] = rsqrtf(var + 1e-5f);
  }
  __syncthreads();
  float mu = red[16], rstd = red[17];
  for (int i = threadIdx.x; i < D; i += 256)
    out[row * D + i] = (__bf16)((x[i] - mu) * rstd * g[i] + bb[i]);
}

// ---------------------------------------------------------------------------
// Launch: full block = QKVproj -> LN+RoPE -> flash-attn(+res) -> LN -> FFN(+res)
// ---------------------------------------------------------------------------
extern "C" void kernel_launch(void* const* d_in, const int* in_sizes, int n_in,
                              void* d_out, int out_size, void* d_ws, size_t ws_size,
                              hipStream_t stream) {
  constexpr int B = 2, N = 2048, M = 2048, D = 1024, H = 16, DFF = 4096;
  constexpr size_t MB = 1024ull * 1024ull;

  const float* query = (const float*)d_in[0];
  const float* ctx   = (const float*)d_in[1];
  const int*   qpos  = (const int*)d_in[2];
  const int*   cpos  = (const int*)d_in[3];
  const float* Wq = (const float*)d_in[4];  const float* bq = (const float*)d_in[5];
  const float* Wk = (const float*)d_in[6];  const float* bk = (const float*)d_in[7];
  const float* Wv = (const float*)d_in[8];  const float* bv = (const float*)d_in[9];
  const float* qn_g = (const float*)d_in[10]; const float* qn_b = (const float*)d_in[11];
  const float* kn_g = (const float*)d_in[12]; const float* kn_b = (const float*)d_in[13];
  const float* fn_g = (const float*)d_in[14]; const float* fn_b = (const float*)d_in[15];
  const float* W1 = (const float*)d_in[16]; const float* b1 = (const float*)d_in[17];
  const float* W2 = (const float*)d_in[18]; const float* b2 = (const float*)d_in[19];
  float* out = (float*)d_out;

  char* ws = (char*)d_ws;
  __bf16* wq_bf = (__bf16*)(ws + 0 * MB);     // 2 MB
  __bf16* wk_bf = (__bf16*)(ws + 2 * MB);     // 2 MB
  __bf16* wv_bf = (__bf16*)(ws + 4 * MB);     // 2 MB
  __bf16* w1_bf = (__bf16*)(ws + 6 * MB);     // 8 MB
  __bf16* w2_bf = (__bf16*)(ws + 14 * MB);    // 8 MB
  __bf16* x_bf  = (__bf16*)(ws + 22 * MB);    // 8 MB
  __bf16* c_bf  = (__bf16*)(ws + 30 * MB);    // 8 MB
  float*  Qf    = (float*) (ws + 38 * MB);    // 16 MB
  float*  Kf    = (float*) (ws + 54 * MB);    // 16 MB
  __bf16* Vb    = (__bf16*)(ws + 70 * MB);    // 8 MB  [B,M,D] bf16
  __bf16* Qb    = (__bf16*)(ws + 78 * MB);    // 8 MB  [B,H,N,64]
  __bf16* Kb    = (__bf16*)(ws + 86 * MB);    // 8 MB  [B,H,M,64]
  float*  Oo    = (float*) (ws + 94 * MB);    // 16 MB [B,N,D] (attn + residual)
  __bf16* hn_bf = (__bf16*)(ws + 110 * MB);   // 8 MB
  __bf16* h1_bf = (__bf16*)(ws + 118 * MB);   // 32 MB

  // 1) conversions to bf16
  cvt_f32_bf16_kernel<<<D * D / 1024, 256, 0, stream>>>(Wq, wq_bf, D * D);
  cvt_f32_bf16_kernel<<<D * D / 1024, 256, 0, stream>>>(Wk, wk_bf, D * D);
  cvt_f32_bf16_kernel<<<D * D / 1024, 256, 0, stream>>>(Wv, wv_bf, D * D);
  cvt_f32_bf16_kernel<<<D * DFF / 1024, 256, 0, stream>>>(W1, w1_bf, D * DFF);
  cvt_f32_bf16_kernel<<<DFF * D / 1024, 256, 0, stream>>>(W2, w2_bf, DFF * D);
  cvt_f32_bf16_kernel<<<B * N * D / 1024, 256, 0, stream>>>(query, x_bf, B * N * D);
  cvt_f32_bf16_kernel<<<B * M * D / 1024, 256, 0, stream>>>(ctx, c_bf, B * M * D);

  // 2) projections (WMMA bf16 GEMMs, TDM-staged A tiles)
  dim3 gD(B * N / 64, D / 64);
  gemm_bf16_kernel<0, 0><<<gD, 256, 0, stream>>>(x_bf, wq_bf, bq, nullptr, Qf, B * N, D, D);
  gemm_bf16_kernel<0, 0><<<gD, 256, 0, stream>>>(c_bf, wk_bf, bk, nullptr, Kf, B * M, D, D);
  gemm_bf16_kernel<0, 1><<<gD, 256, 0, stream>>>(c_bf, wv_bf, bv, nullptr, Vb, B * M, D, D);

  // 3) per-head LN + RoPE (Q carries the 1/sqrt(64) score scale)
  lnrope_kernel<<<B * N * H / 8, 256, 0, stream>>>(Qf, qpos, qn_g, qn_b, Qb, N, 0.125f);
  lnrope_kernel<<<B * M * H / 8, 256, 0, stream>>>(Kf, cpos, kn_g, kn_b, Kb, M, 1.0f);

  // 4) flash attention + residual -> Oo
  dim3 gA(N / 16, H, B);
  flash_attn_kernel<<<gA, 32, 0, stream>>>(Qb, Kb, Vb, query, Oo, N, M);

  // 5) FFN pre-norm
  ln_d_kernel<<<B * N, 256, 0, stream>>>(Oo, fn_g, fn_b, hn_bf, D);

  // 6) FFN: GELU(h*W1+b1) -> h1 (bf16), then h1*W2+b2 + Oo -> out (f32)
  dim3 gU(B * N / 64, DFF / 64);
  gemm_bf16_kernel<1, 1><<<gU, 256, 0, stream>>>(hn_bf, w1_bf, b1, nullptr, h1_bf, B * N, D, DFF);
  dim3 gO(B * N / 64, D / 64);
  gemm_bf16_kernel<0, 0><<<gO, 256, 0, stream>>>(h1_bf, w2_bf, b2, Oo, out, B * N, DFF, D);
}
